// Attention_29326036697501
// MI455X (gfx1250) — matrix operand order repaired and checked by reading, not compiled, not run on hardware
//
#include <hip/hip_runtime.h>
#include <hip/hip_bf16.h>

// ---------------------------------------------------------------------------
// Types for CDNA5 WMMA (gfx1250, wave32)
// ---------------------------------------------------------------------------
typedef __attribute__((ext_vector_type(16))) __bf16 v16bf;
typedef __attribute__((ext_vector_type(8)))  float  v8f;

// fp32 -> bf16 round-to-nearest-even (bit trick)
__device__ __forceinline__ unsigned short f2bf(float f) {
    unsigned int u = __float_as_uint(f);
    u += 0x7FFFu + ((u >> 16) & 1u);
    return (unsigned short)(u >> 16);
}

// ---------------------------------------------------------------------------
// Kernel 1: token-shift mixing + cast to bf16 (xk, xv, xr)
// ---------------------------------------------------------------------------
__global__ __launch_bounds__(256) void mix_cast_kernel(
    const float* __restrict__ x, const float* __restrict__ last_x,
    const float* __restrict__ tmk, const float* __restrict__ tmv,
    const float* __restrict__ tmr,
    unsigned short* __restrict__ xk, unsigned short* __restrict__ xv,
    unsigned short* __restrict__ xr, int T, int D, long long total)
{
    long long idx = (long long)blockIdx.x * blockDim.x + threadIdx.x;
    if (idx >= total) return;
    int d = (int)(idx % D);
    long long bt = idx / D;
    int t = (int)(bt % T);
    long long b = bt / T;

    float xc = x[idx];
    float xs = (t == 0) ? last_x[b * D + d] : x[idx - D];

    float mk = tmk[d], mv = tmv[d], mr = tmr[d];
    xk[idx] = f2bf(xc * mk + xs * (1.0f - mk));
    xv[idx] = f2bf(xc * mv + xs * (1.0f - mv));
    xr[idx] = f2bf(xc * mr + xs * (1.0f - mr));
}

// ---------------------------------------------------------------------------
// Kernel 2: fp32 -> bf16 weight cast
// ---------------------------------------------------------------------------
__global__ __launch_bounds__(256) void cast_bf16_kernel(
    const float* __restrict__ src, unsigned short* __restrict__ dst, int n)
{
    int i = blockIdx.x * blockDim.x + threadIdx.x;
    if (i < n) dst[i] = f2bf(src[i]);
}

// ---------------------------------------------------------------------------
// Kernel 3: NT GEMM  C[M,N] (fp32) = A[M,K] (bf16) @ W[N,K]^T (bf16)
//   Block: 256 threads = 8 waves. Block tile 256(M) x 128(N), BK=32.
//   Waves laid out 4(M) x 2(N); each wave computes 64x64 via 4x4 WMMA tiles
//   (16 v_wmma_f32_16x16x32_bf16 per K-step, fed by 16 ds_load_b128).
//   LDS rows padded to 40 bf16 (80 B) -> conflict-free b128 fragment reads.
//   Double-buffered LDS: one barrier per K-step; staging of tile k+1
//   overlaps WMMA on tile k. global_prefetch for tile k+2.
// ---------------------------------------------------------------------------
#define BM 256
#define BN 128
#define BK 32
#define BKP 40   // padded LDS row length in bf16 elements (80 bytes)

__device__ __forceinline__ void stage_tiles(
    const unsigned short* __restrict__ A,
    const unsigned short* __restrict__ W,
    unsigned short* __restrict__ sA,
    unsigned short* __restrict__ sB,
    int bm, int bn, int K, int k0, int tid)
{
    // A tile: 256 rows x 32 cols; thread tid copies one full row (64 B).
    {
        const uint4* g = (const uint4*)(A + (long long)(bm + tid) * K + k0);
        uint4* d = (uint4*)(sA + tid * BKP);
        d[0] = g[0]; d[1] = g[1]; d[2] = g[2]; d[3] = g[3];
    }
    // W tile: 128 rows x 32 cols; two threads per row (32 B each).
    {
        int row = tid >> 1;
        int col = (tid & 1) << 4;
        const uint4* g = (const uint4*)(W + (long long)(bn + row) * K + k0 + col);
        uint4* d = (uint4*)(sB + row * BKP + col);
        d[0] = g[0]; d[1] = g[1];
    }
}

__global__ __launch_bounds__(256) void gemm_bf16_nt(
    const unsigned short* __restrict__ A,   // [M,K] bf16 bits
    const unsigned short* __restrict__ W,   // [N,K] bf16 bits
    float* __restrict__ C,                  // [M,N] fp32
    int M, int N, int K)
{
    __shared__ __align__(16) unsigned short lA[2][BM * BKP];  // 2 x 20 KB
    __shared__ __align__(16) unsigned short lB[2][BN * BKP];  // 2 x 10 KB

    const int tid   = threadIdx.x;
    const int lane  = tid & 31;           // wave32
    const int wid   = tid >> 5;           // 0..7
    const int waveM = wid & 3;            // 0..3 -> 64 rows each
    const int waveN = wid >> 2;           // 0..1 -> 64 cols each

    const int bn = blockIdx.x * BN;
    const int bm = blockIdx.y * BM;

    const int half = lane >> 4;           // 0 | 1
    const int l15  = lane & 15;

    v8f acc[4][4];
    {
        v8f z = {0.f, 0.f, 0.f, 0.f, 0.f, 0.f, 0.f, 0.f};
#pragma unroll
        for (int i = 0; i < 4; ++i)
#pragma unroll
            for (int j = 0; j < 4; ++j) acc[i][j] = z;
    }

    // Preload first tile pair
    stage_tiles(A, W, lA[0], lB[0], bm, bn, K, 0, tid);
    __syncthreads();

    int cur = 0;
    for (int k0 = 0; k0 < K; k0 += BK) {
        // L2 prefetch of the k+2 tiles (global_prefetch_b8)
        if (k0 + 2 * BK < K) {
            __builtin_prefetch(A + (long long)(bm + tid) * K + k0 + 2 * BK, 0, 1);
            __builtin_prefetch(W + (long long)(bn + (tid >> 1)) * K + k0 + 2 * BK, 0, 1);
        }
        // Stage next tile pair into the alternate buffer (overlaps WMMA below)
        if (k0 + BK < K)
            stage_tiles(A, W, lA[cur ^ 1], lB[cur ^ 1], bm, bn, K, k0 + BK, tid);

        // ---- load WMMA fragments from current LDS buffer ----
        // A 16x32 bf16 fragment: half 0 -> K{0..7,16..23}, half 1 -> K{8..15,24..31}
        const unsigned short* sA = lA[cur];
        const unsigned short* sB = lB[cur];
        v16bf af[4], bfr[4];
#pragma unroll
        for (int i = 0; i < 4; ++i) {
            const unsigned short* rp = sA + (waveM * 64 + i * 16 + l15) * BKP + half * 8;
            uint4* d = (uint4*)&af[i];
            d[0] = *(const uint4*)(rp);
            d[1] = *(const uint4*)(rp + 16);
        }
        // B 32x16 bf16 fragment: lane col = lane&15, half 0 -> K{0..7,16..23}, half 1 -> K{8..15,24..31};
        // column n of B == row n of W -> two 8-bf16 chunks of the W row.
#pragma unroll
        for (int j = 0; j < 4; ++j) {
            const unsigned short* rp = sB + (waveN * 64 + j * 16 + l15) * BKP + half * 8;
            uint4* d = (uint4*)&bfr[j];
            d[0] = *(const uint4*)(rp);
            d[1] = *(const uint4*)(rp + 16);
        }

        // ---- 16 WMMAs per K-step ----
#pragma unroll
        for (int i = 0; i < 4; ++i)
#pragma unroll
            for (int j = 0; j < 4; ++j)
                acc[i][j] = __builtin_amdgcn_wmma_f32_16x16x32_bf16(
                    /*neg_a=*/false, af[i],
                    /*neg_b=*/false, bfr[j],
                    /*c_mod=*/(short)0, acc[i][j],
                    /*reuse_a=*/false, /*reuse_b=*/false);

        __syncthreads();
        cur ^= 1;
    }

    // ---- epilogue: C/D layout VGPR r -> M = r + 8*half, N = lane&15 ----
#pragma unroll
    for (int i = 0; i < 4; ++i) {
        long long gm = (long long)bm + waveM * 64 + i * 16 + half * 8;
#pragma unroll
        for (int j = 0; j < 4; ++j) {
            int gn = bn + waveN * 64 + j * 16 + l15;
            float* cp = C + gm * N + gn;
#pragma unroll
            for (int r = 0; r < 8; ++r)
                cp[(long long)r * N] = acc[i][j][r];
        }
    }
}

// ---------------------------------------------------------------------------
// Kernel 4: WKV sequential scan (fp32) + sigmoid(r) gate, emit bf16
//   One thread per (b,d) channel; consecutive threads -> consecutive d
//   so all loads/stores are coalesced each step.
// ---------------------------------------------------------------------------
__global__ __launch_bounds__(256) void wkv_scan_kernel(
    const float* __restrict__ kbuf, const float* __restrict__ vbuf,
    const float* __restrict__ rbuf,
    const float* __restrict__ time_decay, const float* __restrict__ time_first,
    const float* __restrict__ last_num, const float* __restrict__ last_den,
    unsigned short* __restrict__ abuf, int Bc, int T, int D)
{
    int idx = blockIdx.x * blockDim.x + threadIdx.x;
    if (idx >= Bc * D) return;
    int d = idx % D;
    int b = idx / D;

    const float ew  = __expf(-__expf(time_decay[d]));
    const float tf  = time_first[d];
    float num = last_num[(long long)b * D + d];
    float den = last_den[(long long)b * D + d];

    long long base = (long long)b * T * D + d;
    for (int t = 0; t < T; ++t) {
        long long off = base + (long long)t * D;
        float kk = kbuf[off];
        float vv = vbuf[off];
        float rr = rbuf[off];
        float ek  = __expf(kk);
        float euk = __expf(tf + kk);
        float out = (num + euk * vv) / (den + euk);
        num = ew * num + ek * vv;
        den = ew * den + ek;
        float sr = 1.0f / (1.0f + __expf(-rr));
        abuf[off] = f2bf(out * sr);
    }
}

// ---------------------------------------------------------------------------
// Host-side orchestration
// ---------------------------------------------------------------------------
extern "C" void kernel_launch(void* const* d_in, const int* in_sizes, int n_in,
                              void* d_out, int out_size, void* d_ws, size_t ws_size,
                              hipStream_t stream)
{
    (void)in_sizes; (void)n_in; (void)out_size; (void)ws_size;

    const int Bc = 8, T = 2048, D = 1024;
    const int M = Bc * T;                       // 16384
    const long long MT = (long long)M * D;      // 16,777,216
    const long long DD = (long long)D * D;      // 1,048,576

    // Inputs (setup_inputs order)
    const float* x    = (const float*)d_in[0];
    const float* lx   = (const float*)d_in[1];
    const float* lnum = (const float*)d_in[2];
    const float* lden = (const float*)d_in[3];
    const float* tdec = (const float*)d_in[4];
    const float* tfst = (const float*)d_in[5];
    const float* tmk  = (const float*)d_in[6];
    const float* tmv  = (const float*)d_in[7];
    const float* tmr  = (const float*)d_in[8];
    const float* Wk   = (const float*)d_in[9];
    const float* Wv   = (const float*)d_in[10];
    const float* Wr   = (const float*)d_in[11];
    const float* Wo   = (const float*)d_in[12];

    // Workspace layout (256B-aligned slabs)
    char* ws = (char*)d_ws;
    size_t off = 0;
    auto alloc = [&](size_t bytes) -> char* {
        char* p = ws + off;
        off = (off + bytes + 255) & ~(size_t)255;
        return p;
    };
    unsigned short* xk  = (unsigned short*)alloc(MT * 2);   // reused as abuf later
    unsigned short* xv  = (unsigned short*)alloc(MT * 2);
    unsigned short* xr  = (unsigned short*)alloc(MT * 2);
    unsigned short* Wkb = (unsigned short*)alloc(DD * 2);
    unsigned short* Wvb = (unsigned short*)alloc(DD * 2);
    unsigned short* Wrb = (unsigned short*)alloc(DD * 2);
    unsigned short* Wob = (unsigned short*)alloc(DD * 2);
    float* kbuf = (float*)alloc(MT * 4);
    float* vbuf = (float*)alloc(MT * 4);
    float* rbuf = (float*)alloc(MT * 4);
    unsigned short* abuf = xk;   // xk is dead after GEMM-K; safe alias

    // 1) mix + cast
    {
        long long total = MT;
        int blocks = (int)((total + 255) / 256);
        mix_cast_kernel<<<blocks, 256, 0, stream>>>(x, lx, tmk, tmv, tmr,
                                                    xk, xv, xr, T, D, total);
    }

    // 2) weight casts
    {
        int n = (int)DD;
        int blocks = (n + 255) / 256;
        cast_bf16_kernel<<<blocks, 256, 0, stream>>>(Wk, Wkb, n);
        cast_bf16_kernel<<<blocks, 256, 0, stream>>>(Wv, Wvb, n);
        cast_bf16_kernel<<<blocks, 256, 0, stream>>>(Wr, Wrb, n);
        cast_bf16_kernel<<<blocks, 256, 0, stream>>>(Wo, Wob, n);
    }

    // 3) K/V/R projections (WMMA bf16 -> fp32)
    {
        dim3 grid(D / BN, M / BM);   // (8, 64)
        gemm_bf16_nt<<<grid, 256, 0, stream>>>(xk, Wkb, kbuf, M, D, D);
        gemm_bf16_nt<<<grid, 256, 0, stream>>>(xv, Wvb, vbuf, M, D, D);
        gemm_bf16_nt<<<grid, 256, 0, stream>>>(xr, Wrb, rbuf, M, D, D);
    }

    // 4) WKV scan + sigmoid gate (fp32 recurrence, bf16 output)
    {
        int n = Bc * D;
        wkv_scan_kernel<<<(n + 255) / 256, 256, 0, stream>>>(
            kbuf, vbuf, rbuf, tdec, tfst, lnum, lden, abuf, Bc, T, D);
    }

    // 5) output projection -> d_out (fp32)
    {
        dim3 grid(D / BN, M / BM);
        gemm_bf16_nt<<<grid, 256, 0, stream>>>(abuf, Wob, (float*)d_out, M, D, D);
    }
}
